// Rwkv_738734375128
// MI455X (gfx1250) — matrix operand order repaired and checked by reading, not compiled, not run on hardware
//
#include <hip/hip_runtime.h>

// ---------------------------------------------------------------------------
// RWKV forward for MI455X (gfx1250). All GEMMs run on WMMA bf16 matrix cores
// (v_wmma_f32_16x16x32_bf16) with tiles staged via CDNA5 async global->LDS
// copies (global_load_async_to_lds_b128, ASYNCcnt), double-buffered LDS with
// a 2-tile software pipeline (buffer index is a compile-time constant).
// ---------------------------------------------------------------------------

#define DD   1024
#define FFD  4096
#define LL   4
#define VV   50257
#define BT   2048          // B*T = 2*1024

#define BM 128
#define BN 128
#define BK 32

typedef __attribute__((ext_vector_type(8)))  unsigned short v8u;
typedef __attribute__((ext_vector_type(16))) unsigned short v16u;
typedef __attribute__((ext_vector_type(16))) __bf16         v16bf;
typedef __attribute__((ext_vector_type(8)))  float          v8f;

__device__ __forceinline__ unsigned short f2bf(float f) {
    unsigned int u = __float_as_uint(f);
    u += 0x7fffu + ((u >> 16) & 1u);          // round-to-nearest-even
    return (unsigned short)(u >> 16);
}

__device__ __forceinline__ float sigmf(float x) {
    return 1.0f / (1.0f + expf(-x));
}

// ---------------------------------------------------------------------------
// fp32 -> bf16 bit-pattern conversion
// ---------------------------------------------------------------------------
__global__ __launch_bounds__(256) void f32_to_bf16_k(const float* __restrict__ s,
                                                     unsigned short* __restrict__ d,
                                                     size_t n) {
    size_t i = (size_t)blockIdx.x * 256 + threadIdx.x;
    if (i < n) d[i] = f2bf(s[i]);
}

// ---------------------------------------------------------------------------
// Generic WMMA bf16 GEMM:  C[M,N] = A[M,K] * W[N,K]^T   (fp32 accumulate)
// Requires M % 128 == 0, K % 64 == 0. N guarded at the epilogue only:
// B-tile row indices are clamped to N-1 (never stored columns may hold
// garbage accumulations; they are finite and discarded).
//
// Pipeline (4 async-load instructions per wave per tile, ASYNCcnt += 4):
//   prologue: tile0 -> buf0, tile1 -> buf1            (8 in flight)
//   loop:     wait<=4 (buf0 done), compute buf0,
//             refill buf0 (tile k+2), wait<=4 (buf1 done), compute buf1,
//             refill buf1 (tile k+3)
// ---------------------------------------------------------------------------
__global__ __launch_bounds__(256) void gemm_bf16_k(const unsigned short* __restrict__ A,
                                                   const unsigned short* __restrict__ W,
                                                   float* __restrict__ C,
                                                   int M, int N, int K) {
    __shared__ unsigned short As[2][BM * BK];
    __shared__ unsigned short Bs[2][BN * BK];

    const int t    = threadIdx.x;
    const int lane = t & 31;
    const int wid  = t >> 5;
    const int wrow = wid & 3;        // 4 wave-rows of 32
    const int wcol = wid >> 2;       // 2 wave-cols of 64
    const int rowBase = blockIdx.y * BM;
    const int colBase = blockIdx.x * BN;

    v8f acc[2][4];
    for (int i = 0; i < 2; ++i)
        for (int j = 0; j < 4; ++j)
            acc[i][j] = (v8f){};

    const int lr = t >> 1;           // 0..127 tile row handled by this thread
    const int lc = (t & 1) << 4;     // 0 or 16 (half-row of 32)
    const int nIdx = colBase + lr;
    const int nCl  = (nIdx < N) ? nIdx : (N - 1);   // clamp: keep EXEC uniform

    const unsigned short* gA = A + (size_t)(rowBase + lr) * K + lc;
    const unsigned short* gW = W + (size_t)nCl * K + lc;

    // Per-lane LDS byte addresses (low 32 bits of the flat LDS address are the
    // LDS offset per the CDNA5 aperture mapping).
    unsigned aLds[2], bLds[2];
    aLds[0] = (unsigned)(unsigned long long)&As[0][lr * BK + lc];
    aLds[1] = (unsigned)(unsigned long long)&As[1][lr * BK + lc];
    bLds[0] = (unsigned)(unsigned long long)&Bs[0][lr * BK + lc];
    bLds[1] = (unsigned)(unsigned long long)&Bs[1][lr * BK + lc];

#define ISSUE_TILE(K0, BUF)                                                     \
    do {                                                                        \
        const unsigned short* pa = gA + (K0);                                   \
        const unsigned short* pw = gW + (K0);                                   \
        asm volatile("global_load_async_to_lds_b128 %0, %1, off"                \
                     :: "v"(aLds[BUF]), "v"(pa) : "memory");                    \
        asm volatile("global_load_async_to_lds_b128 %0, %1, off offset:16"      \
                     :: "v"(aLds[BUF]), "v"(pa) : "memory");                    \
        asm volatile("global_load_async_to_lds_b128 %0, %1, off"                \
                     :: "v"(bLds[BUF]), "v"(pw) : "memory");                    \
        asm volatile("global_load_async_to_lds_b128 %0, %1, off offset:16"      \
                     :: "v"(bLds[BUF]), "v"(pw) : "memory");                    \
    } while (0)

    // Fragment lane mapping (contiguous k = 16*(lane/16) + s, same order for A and B):
    const int mrow  = lane & 15;
    const int koffA = (lane >> 4) << 4;   // A: K 0..15 / 16..31 per half-wave
    const int kbB   = (lane >> 4) << 4;   // B: K 0..15 / 16..31 per half-wave

#define COMPUTE_TILE(BUF)                                                       \
    do {                                                                        \
        v16bf afrag[2], bfrag[4];                                               \
        for (int i = 0; i < 2; ++i) {                                           \
            const int r0 = wrow * 32 + i * 16 + mrow;                           \
            v8u lo = *(const v8u*)&As[BUF][r0 * BK + koffA];                    \
            v8u hi = *(const v8u*)&As[BUF][r0 * BK + 8 + koffA];                \
            v16u ar = __builtin_shufflevector(lo, hi,                           \
                        0, 1, 2, 3, 4, 5, 6, 7, 8, 9, 10, 11, 12, 13, 14, 15); \
            afrag[i] = __builtin_bit_cast(v16bf, ar);                           \
        }                                                                       \
        for (int j = 0; j < 4; ++j) {                                           \
            const int c0 = wcol * 64 + j * 16 + (lane & 15);                    \
            v16u br = *(const v16u*)&Bs[BUF][c0 * BK + kbB];                    \
            bfrag[j] = __builtin_bit_cast(v16bf, br);                           \
        }                                                                       \
        for (int i = 0; i < 2; ++i)                                             \
            for (int j = 0; j < 4; ++j)                                         \
                acc[i][j] = __builtin_amdgcn_wmma_f32_16x16x32_bf16(            \
                    false, afrag[i], false, bfrag[j],                           \
                    (short)0, acc[i][j], false, false);                         \
    } while (0)

    ISSUE_TILE(0, 0);
    if (BK < K) ISSUE_TILE(BK, 1);

    for (int k0 = 0; k0 < K; k0 += 2 * BK) {
        // ---- tile in buf0 (k0) ----
        if (k0 + BK < K) {
            asm volatile("s_wait_asynccnt 0x4" ::: "memory");  // buf0 complete
        } else {
            asm volatile("s_wait_asynccnt 0x0" ::: "memory");
        }
        __syncthreads();
        COMPUTE_TILE(0);
        __syncthreads();                        // reads of buf0 done everywhere
        if (k0 + 2 * BK < K) {
            ISSUE_TILE(k0 + 2 * BK, 0);         // refill buf0 with tile k+2
            asm volatile("s_wait_asynccnt 0x4" ::: "memory");  // buf1 complete
        } else {
            asm volatile("s_wait_asynccnt 0x0" ::: "memory");
        }

        // ---- tile in buf1 (k0 + BK) ----
        __syncthreads();
        COMPUTE_TILE(1);
        __syncthreads();                        // reads of buf1 done everywhere
        if (k0 + 3 * BK < K)
            ISSUE_TILE(k0 + 3 * BK, 1);         // refill buf1 with tile k+3
    }
#undef ISSUE_TILE
#undef COMPUTE_TILE

    // ---- epilogue: C layout VGPR v -> M = v + 8*(lane>=16), N = lane&15 ----
    const int nn   = lane & 15;
    const int mofs = (lane >> 4) << 3;
    for (int i = 0; i < 2; ++i)
        for (int j = 0; j < 4; ++j) {
            const int gn = colBase + wcol * 64 + j * 16 + nn;
            if (gn < N) {
                const int gm = rowBase + wrow * 32 + i * 16 + mofs;
                for (int v = 0; v < 8; ++v)
                    C[(size_t)(gm + v) * N + gn] = acc[i][j][v];
            }
        }
}

// ---------------------------------------------------------------------------
// Block-wide sum reduction helper (256 threads)
// ---------------------------------------------------------------------------
__device__ __forceinline__ float block_sum(float v, float* red) {
    const int t = threadIdx.x;
    red[t] = v;
    __syncthreads();
    for (int s = 128; s > 0; s >>= 1) {
        if (t < s) red[t] += red[t + s];
        __syncthreads();
    }
    float r = red[0];
    __syncthreads();
    return r;
}

// ---------------------------------------------------------------------------
// Embedding gather + LayerNorm(ln0) -> x (fp32).  One block per token row.
// ---------------------------------------------------------------------------
__global__ __launch_bounds__(256) void gather_ln_k(const int* __restrict__ tokens,
                                                   const float* __restrict__ emb,
                                                   const float* __restrict__ w,
                                                   const float* __restrict__ b,
                                                   float* __restrict__ x) {
    __shared__ float red[256];
    const int row = blockIdx.x, t = threadIdx.x;
    const float* e = emb + (size_t)tokens[row] * DD;
    float vals[4], s = 0.f;
    for (int j = 0; j < 4; ++j) { vals[j] = e[t + j * 256]; s += vals[j]; }
    const float mu = block_sum(s, red) * (1.f / DD);
    float s2 = 0.f;
    for (int j = 0; j < 4; ++j) { float d = vals[j] - mu; s2 += d * d; }
    const float rs = rsqrtf(block_sum(s2, red) * (1.f / DD) + 1e-5f);
    for (int j = 0; j < 4; ++j) {
        const int c = t + j * 256;
        x[(size_t)row * DD + c] = (vals[j] - mu) * rs * w[c] + b[c];
    }
}

// ---------------------------------------------------------------------------
// LN + time-mix: produce xk/xv/xr (bf16) for the attention GEMMs.
// ---------------------------------------------------------------------------
__global__ __launch_bounds__(256) void ln_mix3_k(const float* __restrict__ x,
                                                 const float* __restrict__ w,
                                                 const float* __restrict__ b,
                                                 const float* __restrict__ mk,
                                                 const float* __restrict__ mv,
                                                 const float* __restrict__ mr,
                                                 const float* __restrict__ sx,
                                                 unsigned short* __restrict__ ok,
                                                 unsigned short* __restrict__ ov,
                                                 unsigned short* __restrict__ orr) {
    __shared__ float red[256];
    const int row = blockIdx.x, t = threadIdx.x;
    const float* xr = x + (size_t)row * DD;
    float vals[4], s = 0.f;
    for (int j = 0; j < 4; ++j) { vals[j] = xr[t + j * 256]; s += vals[j]; }
    const float mu = block_sum(s, red) * (1.f / DD);
    float s2 = 0.f;
    for (int j = 0; j < 4; ++j) { float d = vals[j] - mu; s2 += d * d; }
    const float rs = rsqrtf(block_sum(s2, red) * (1.f / DD) + 1e-5f);
    for (int j = 0; j < 4; ++j) {
        const int c = t + j * 256;
        const float h = (vals[j] - mu) * rs * w[c] + b[c];
        const float sv = sx[c];
        float m;
        m = mk[c]; ok [(size_t)row * DD + c] = f2bf(h * m + sv * (1.f - m));
        m = mv[c]; ov [(size_t)row * DD + c] = f2bf(h * m + sv * (1.f - m));
        m = mr[c]; orr[(size_t)row * DD + c] = f2bf(h * m + sv * (1.f - m));
    }
}

// LN + channel-mix (two outputs)
__global__ __launch_bounds__(256) void ln_mix2_k(const float* __restrict__ x,
                                                 const float* __restrict__ w,
                                                 const float* __restrict__ b,
                                                 const float* __restrict__ mk,
                                                 const float* __restrict__ mr,
                                                 const float* __restrict__ sx,
                                                 unsigned short* __restrict__ ok,
                                                 unsigned short* __restrict__ orr) {
    __shared__ float red[256];
    const int row = blockIdx.x, t = threadIdx.x;
    const float* xr = x + (size_t)row * DD;
    float vals[4], s = 0.f;
    for (int j = 0; j < 4; ++j) { vals[j] = xr[t + j * 256]; s += vals[j]; }
    const float mu = block_sum(s, red) * (1.f / DD);
    float s2 = 0.f;
    for (int j = 0; j < 4; ++j) { float d = vals[j] - mu; s2 += d * d; }
    const float rs = rsqrtf(block_sum(s2, red) * (1.f / DD) + 1e-5f);
    for (int j = 0; j < 4; ++j) {
        const int c = t + j * 256;
        const float h = (vals[j] - mu) * rs * w[c] + b[c];
        const float sv = sx[c];
        float m;
        m = mk[c]; ok [(size_t)row * DD + c] = f2bf(h * m + sv * (1.f - m));
        m = mr[c]; orr[(size_t)row * DD + c] = f2bf(h * m + sv * (1.f - m));
    }
}

// Final LN -> bf16 activations for the head GEMM
__global__ __launch_bounds__(256) void ln_out_k(const float* __restrict__ x,
                                                const float* __restrict__ w,
                                                const float* __restrict__ b,
                                                unsigned short* __restrict__ o) {
    __shared__ float red[256];
    const int row = blockIdx.x, t = threadIdx.x;
    const float* xr = x + (size_t)row * DD;
    float vals[4], s = 0.f;
    for (int j = 0; j < 4; ++j) { vals[j] = xr[t + j * 256]; s += vals[j]; }
    const float mu = block_sum(s, red) * (1.f / DD);
    float s2 = 0.f;
    for (int j = 0; j < 4; ++j) { float d = vals[j] - mu; s2 += d * d; }
    const float rs = rsqrtf(block_sum(s2, red) * (1.f / DD) + 1e-5f);
    for (int j = 0; j < 4; ++j) {
        const int c = t + j * 256;
        o[(size_t)row * DD + c] = f2bf((vals[j] - mu) * rs * w[c] + b[c]);
    }
}

// ---------------------------------------------------------------------------
// WKV + receptance gate:  a = sigmoid(r) * (num + e^(tf+k) * v)/(den + e^(tf+k))
// ---------------------------------------------------------------------------
__global__ __launch_bounds__(256) void wkv_k(const float* __restrict__ k,
                                             const float* __restrict__ v,
                                             const float* __restrict__ r,
                                             const float* __restrict__ tf,
                                             const float* __restrict__ num,
                                             const float* __restrict__ den,
                                             unsigned short* __restrict__ a, int n) {
    const int i = blockIdx.x * 256 + threadIdx.x;
    if (i >= n) return;
    const int d = i & (DD - 1);
    const float ek  = expf(tf[d] + k[i]);
    const float wkv = (num[d] + ek * v[i]) / (den[d] + ek);
    a[i] = f2bf(sigmf(r[i]) * wkv);
}

__global__ __launch_bounds__(256) void add_k(float* __restrict__ x,
                                             const float* __restrict__ c, int n) {
    const int i = blockIdx.x * 256 + threadIdx.x;
    if (i < n) x[i] += c[i];
}

__global__ __launch_bounds__(256) void relusq_k(const float* __restrict__ k2,
                                                unsigned short* __restrict__ o, int n) {
    const int i = blockIdx.x * 256 + threadIdx.x;
    if (i >= n) return;
    const float t = fmaxf(k2[i], 0.f);
    o[i] = f2bf(t * t);
}

__global__ __launch_bounds__(256) void gateadd_k(float* __restrict__ x,
                                                 const float* __restrict__ r2,
                                                 const float* __restrict__ vk, int n) {
    const int i = blockIdx.x * 256 + threadIdx.x;
    if (i < n) x[i] += sigmf(r2[i]) * vk[i];
}

// ---------------------------------------------------------------------------
// Global softmax over all BT*V logits (3 passes), in place on d_out.
// ---------------------------------------------------------------------------
__global__ __launch_bounds__(256) void reduce_max_k(const float* __restrict__ x, size_t n,
                                                    float* __restrict__ partial) {
    __shared__ float red[256];
    float m = -3.4e38f;
    for (size_t i = (size_t)blockIdx.x * 256 + threadIdx.x; i < n;
         i += (size_t)gridDim.x * 256)
        m = fmaxf(m, x[i]);
    red[threadIdx.x] = m;
    __syncthreads();
    for (int s = 128; s > 0; s >>= 1) {
        if (threadIdx.x < s) red[threadIdx.x] = fmaxf(red[threadIdx.x], red[threadIdx.x + s]);
        __syncthreads();
    }
    if (threadIdx.x == 0) partial[blockIdx.x] = red[0];
}

__global__ __launch_bounds__(256) void final_max_k(const float* __restrict__ partial, int nb,
                                                   float* __restrict__ scal) {
    __shared__ float red[256];
    float m = -3.4e38f;
    for (int i = threadIdx.x; i < nb; i += 256) m = fmaxf(m, partial[i]);
    red[threadIdx.x] = m;
    __syncthreads();
    for (int s = 128; s > 0; s >>= 1) {
        if (threadIdx.x < s) red[threadIdx.x] = fmaxf(red[threadIdx.x], red[threadIdx.x + s]);
        __syncthreads();
    }
    if (threadIdx.x == 0) scal[0] = red[0];
}

__global__ __launch_bounds__(256) void reduce_sumexp_k(const float* __restrict__ x, size_t n,
                                                       const float* __restrict__ scal,
                                                       float* __restrict__ partial) {
    __shared__ float red[256];
    const float mx = scal[0];
    float s = 0.f;
    for (size_t i = (size_t)blockIdx.x * 256 + threadIdx.x; i < n;
         i += (size_t)gridDim.x * 256)
        s += expf(x[i] - mx);
    red[threadIdx.x] = s;
    __syncthreads();
    for (int st = 128; st > 0; st >>= 1) {
        if (threadIdx.x < st) red[threadIdx.x] += red[threadIdx.x + st];
        __syncthreads();
    }
    if (threadIdx.x == 0) partial[blockIdx.x] = red[0];
}

__global__ __launch_bounds__(256) void final_sum_k(const float* __restrict__ partial, int nb,
                                                   float* __restrict__ scal) {
    __shared__ float red[256];
    float s = 0.f;
    for (int i = threadIdx.x; i < nb; i += 256) s += partial[i];
    red[threadIdx.x] = s;
    __syncthreads();
    for (int st = 128; st > 0; st >>= 1) {
        if (threadIdx.x < st) red[threadIdx.x] += red[threadIdx.x + st];
        __syncthreads();
    }
    if (threadIdx.x == 0) scal[1] = red[0];
}

__global__ __launch_bounds__(256) void normalize_k(float* __restrict__ x, size_t n,
                                                   const float* __restrict__ scal) {
    const float mx = scal[0];
    const float inv = 1.f / scal[1];
    for (size_t i = (size_t)blockIdx.x * 256 + threadIdx.x; i < n;
         i += (size_t)gridDim.x * 256)
        x[i] = expf(x[i] - mx) * inv;
}

// ---------------------------------------------------------------------------
// Host orchestration
// ---------------------------------------------------------------------------
extern "C" void kernel_launch(void* const* d_in, const int* in_sizes, int n_in,
                              void* d_out, int out_size, void* d_ws, size_t ws_size,
                              hipStream_t stream) {
    (void)in_sizes; (void)n_in; (void)out_size; (void)ws_size;

    const int*   tokens = (const int*)  d_in[0];
    const float* emb    = (const float*)d_in[1];
    const float* ln0_w  = (const float*)d_in[2];
    const float* ln0_b  = (const float*)d_in[3];
    const float* ln1_w  = (const float*)d_in[4];
    const float* ln1_b  = (const float*)d_in[5];
    const float* ln2_w  = (const float*)d_in[6];
    const float* ln2_b  = (const float*)d_in[7];
    /* d_in[8] = att_time_decay (unused by reference) */
    const float* atf    = (const float*)d_in[9];
    const float* amk    = (const float*)d_in[10];
    const float* amv    = (const float*)d_in[11];
    const float* amr    = (const float*)d_in[12];
    const float* aWk    = (const float*)d_in[13];
    const float* aWv    = (const float*)d_in[14];
    const float* aWr    = (const float*)d_in[15];
    const float* aWo    = (const float*)d_in[16];
    const float* fmk    = (const float*)d_in[17];
    const float* fmr    = (const float*)d_in[18];
    const float* fWk    = (const float*)d_in[19];
    const float* fWr    = (const float*)d_in[20];
    const float* fWv    = (const float*)d_in[21];
    const float* lnow   = (const float*)d_in[22];
    const float* lnob   = (const float*)d_in[23];
    const float* headW  = (const float*)d_in[24];
    const float* asx    = (const float*)d_in[25];
    const float* anum   = (const float*)d_in[26];
    const float* aden   = (const float*)d_in[27];
    const float* fsx    = (const float*)d_in[28];

    // ---- workspace arena ----
    char* ws = (char*)d_ws;
    size_t off = 0;
    auto alloc = [&](size_t bytes) -> void* {
        void* p = ws + off;
        off += (bytes + 255) & ~(size_t)255;
        return p;
    };

    unsigned short *bWk[LL], *bWv[LL], *bWr[LL], *bWo[LL], *bfWk[LL], *bfWr[LL], *bfWv[LL];
    for (int i = 0; i < LL; ++i) {
        bWk [i] = (unsigned short*)alloc((size_t)DD * DD * 2);
        bWv [i] = (unsigned short*)alloc((size_t)DD * DD * 2);
        bWr [i] = (unsigned short*)alloc((size_t)DD * DD * 2);
        bWo [i] = (unsigned short*)alloc((size_t)DD * DD * 2);
        bfWk[i] = (unsigned short*)alloc((size_t)FFD * DD * 2);
        bfWr[i] = (unsigned short*)alloc((size_t)DD * DD * 2);
        bfWv[i] = (unsigned short*)alloc((size_t)DD * FFD * 2);
    }
    unsigned short* bHead = (unsigned short*)alloc((size_t)VV * DD * 2);

    float*          x   = (float*)         alloc((size_t)BT * DD * 4);
    unsigned short* xk  = (unsigned short*)alloc((size_t)BT * DD * 2);
    unsigned short* xv  = (unsigned short*)alloc((size_t)BT * DD * 2);
    unsigned short* xr  = (unsigned short*)alloc((size_t)BT * DD * 2);
    float*          kb  = (float*)         alloc((size_t)BT * DD * 4);
    float*          vb  = (float*)         alloc((size_t)BT * DD * 4);
    float*          rb  = (float*)         alloc((size_t)BT * DD * 4);
    unsigned short* ab  = (unsigned short*)alloc((size_t)BT * DD * 2);
    float*          k2  = (float*)         alloc((size_t)BT * FFD * 4);
    unsigned short* k2b = (unsigned short*)alloc((size_t)BT * FFD * 2);
    unsigned short* hb  = (unsigned short*)alloc((size_t)BT * DD * 2);
    float*          part= (float*)         alloc(1024 * 4);
    float*          scal= (float*)         alloc(2 * 4);

    auto conv = [&](const float* s, unsigned short* d, size_t n) {
        f32_to_bf16_k<<<dim3((unsigned)((n + 255) / 256)), dim3(256), 0, stream>>>(s, d, n);
    };

    // ---- convert all GEMM weights to bf16 once ----
    for (int i = 0; i < LL; ++i) {
        conv(aWk + (size_t)i * DD * DD,  bWk[i],  (size_t)DD * DD);
        conv(aWv + (size_t)i * DD * DD,  bWv[i],  (size_t)DD * DD);
        conv(aWr + (size_t)i * DD * DD,  bWr[i],  (size_t)DD * DD);
        conv(aWo + (size_t)i * DD * DD,  bWo[i],  (size_t)DD * DD);
        conv(fWk + (size_t)i * FFD * DD, bfWk[i], (size_t)FFD * DD);
        conv(fWr + (size_t)i * DD * DD,  bfWr[i], (size_t)DD * DD);
        conv(fWv + (size_t)i * DD * FFD, bfWv[i], (size_t)DD * FFD);
    }
    conv(headW, bHead, (size_t)VV * DD);

    // ---- embedding gather + ln0 ----
    gather_ln_k<<<BT, 256, 0, stream>>>(tokens, emb, ln0_w, ln0_b, x);

    const dim3 gD((DD + BN - 1) / BN, BT / BM);    // [2048,1024] output
    const dim3 gF((FFD + BN - 1) / BN, BT / BM);   // [2048,4096] output
    const int  nE = BT * DD;
    const int  nF = BT * FFD;

    for (int i = 0; i < LL; ++i) {
        // ---- time-mix (attention) ----
        ln_mix3_k<<<BT, 256, 0, stream>>>(x, ln1_w + i * DD, ln1_b + i * DD,
                                          amk + i * DD, amv + i * DD, amr + i * DD,
                                          asx + i * DD, xk, xv, xr);
        gemm_bf16_k<<<gD, 256, 0, stream>>>(xk, bWk[i], kb, BT, DD, DD);
        gemm_bf16_k<<<gD, 256, 0, stream>>>(xv, bWv[i], vb, BT, DD, DD);
        gemm_bf16_k<<<gD, 256, 0, stream>>>(xr, bWr[i], rb, BT, DD, DD);
        wkv_k<<<(nE + 255) / 256, 256, 0, stream>>>(kb, vb, rb, atf + i * DD,
                                                    anum + i * DD, aden + i * DD, ab, nE);
        gemm_bf16_k<<<gD, 256, 0, stream>>>(ab, bWo[i], kb, BT, DD, DD);
        add_k<<<(nE + 255) / 256, 256, 0, stream>>>(x, kb, nE);

        // ---- channel-mix (ffn) ----
        ln_mix2_k<<<BT, 256, 0, stream>>>(x, ln2_w + i * DD, ln2_b + i * DD,
                                          fmk + i * DD, fmr + i * DD, fsx + i * DD, xk, xr);
        gemm_bf16_k<<<gF, 256, 0, stream>>>(xk, bfWk[i], k2, BT, FFD, DD);
        gemm_bf16_k<<<gD, 256, 0, stream>>>(xr, bfWr[i], rb, BT, DD, DD);
        relusq_k<<<(nF + 255) / 256, 256, 0, stream>>>(k2, k2b, nF);
        gemm_bf16_k<<<gD, 256, 0, stream>>>(k2b, bfWv[i], vb, BT, DD, FFD);
        gateadd_k<<<(nE + 255) / 256, 256, 0, stream>>>(x, rb, vb, nE);
    }

    // ---- head + global softmax (logits written directly into d_out) ----
    float* out = (float*)d_out;
    ln_out_k<<<BT, 256, 0, stream>>>(x, lnow, lnob, hb);
    const dim3 gH((VV + BN - 1) / BN, BT / BM);
    gemm_bf16_k<<<gH, 256, 0, stream>>>(hb, bHead, out, BT, VV, DD);

    const size_t nOut = (size_t)BT * VV;
    reduce_max_k   <<<1024, 256, 0, stream>>>(out, nOut, part);
    final_max_k    <<<1,    256, 0, stream>>>(part, 1024, scal);
    reduce_sumexp_k<<<1024, 256, 0, stream>>>(out, nOut, scal, part);
    final_sum_k    <<<1,    256, 0, stream>>>(part, 1024, scal);
    normalize_k    <<<4096, 256, 0, stream>>>(out, nOut, scal);
}
